// DistanceLoss_8942121910555
// MI455X (gfx1250) — matrix operand order repaired and checked by reading, not compiled, not run on hardware
//
#include <hip/hip_runtime.h>
#include <hip/hip_bf16.h>
#include <math.h>

typedef __attribute__((ext_vector_type(16))) __bf16        v16bf;
typedef __attribute__((ext_vector_type(8)))  float         v8f;
typedef __attribute__((ext_vector_type(4)))  unsigned int  u32x4;

#define N_ROWS 16384
#define N_COLS 2048
#define DIM    512
#define LDA    (DIM + 8)   // bf16 elems; 1040B row stride -> conflict-free b128 LDS loads
#define MTILES 4
#define MROWS  (MTILES * 16)   // 64 rows per block: 4x B-fragment reuse vs M=16

// ---------------- prep: e2[c] = ||emb_c||^2 ; emb -> bf16 hi/lo ----------------
__global__ __launch_bounds__(256) void prep_emb(const float* __restrict__ emb,
                                                __bf16* __restrict__ Bh,
                                                __bf16* __restrict__ Bl,
                                                float* __restrict__ e2) {
    const int c = blockIdx.x;
    const int tid = threadIdx.x;
    const size_t base = (size_t)c * DIM;
    float v0 = emb[base + tid];
    float v1 = emb[base + tid + 256];

    __shared__ float red[256];
    red[tid] = v0 * v0 + v1 * v1;
    __syncthreads();
    for (int s = 128; s > 0; s >>= 1) {
        if (tid < s) red[tid] += red[tid + s];
        __syncthreads();
    }
    if (tid == 0) e2[c] = red[0];

    __bf16 h0 = (__bf16)v0;
    Bh[base + tid] = h0;
    Bl[base + tid] = (__bf16)(v0 - (float)h0);
    __bf16 h1 = (__bf16)v1;
    Bh[base + tid + 256] = h1;
    Bl[base + tid + 256] = (__bf16)(v1 - (float)h1);
}

// ---------------- main: 64-row tile per block: normalize, WMMA GEMM, fused loss ----------------
__global__ __launch_bounds__(256) void dist_main(const float* __restrict__ WO,
                                                 const int* __restrict__ label,
                                                 const __bf16* __restrict__ Bh,
                                                 const __bf16* __restrict__ Bl,
                                                 const float* __restrict__ e2,
                                                 float* __restrict__ out) {
    __shared__ __bf16 sAh[MROWS][LDA];   // 66,560 B
    __shared__ __bf16 sAl[MROWS][LDA];   // 66,560 B
    __shared__ float  sX2[MROWS];
    __shared__ int    sLb[MROWS];
    __shared__ float  sMin[8][MROWS];
    __shared__ float  sLab[8][MROWS];

    const int tid = threadIdx.x;
    const int rowBase = blockIdx.x * MROWS;

    // ---- stage + normalize A (64 x 512 fp32 -> bf16 hi/lo in LDS), 16 rows/pass ----
    {
        const int g0  = tid >> 4;    // half-wave per row
        const int l16 = tid & 15;
#pragma unroll
        for (int it = 0; it < MTILES; ++it) {
            const int g = g0 + it * 16;
            const float* src = WO + (size_t)(rowBase + g) * DIM;
            float vals[32];
            float ss = 0.f;
#pragma unroll
            for (int j = 0; j < 32; ++j) {
                float v = src[l16 + j * 16];
                vals[j] = v;
                ss += v * v;
            }
#pragma unroll
            for (int m = 8; m >= 1; m >>= 1) ss += __shfl_xor(ss, m, 32);  // within 16-lane half
            float rn = 1.0f / fmaxf(sqrtf(ss), 1e-12f);
            if (l16 == 0) {
                sX2[g] = ss * rn * rn;              // == sum(WO_n^2) in fp32
                sLb[g] = label[rowBase + g];
            }
#pragma unroll
            for (int j = 0; j < 32; ++j) {
                int k = l16 + j * 16;
                float f = vals[j] * rn;
                __bf16 h = (__bf16)f;
                sAh[g][k] = h;
                sAl[g][k] = (__bf16)(f - (float)h);
            }
        }
    }
    __syncthreads();

    const int wid  = tid >> 5;   // wave -> 16-column strip within 128-col chunk
    const int lane = tid & 31;
    const int c16  = lane & 15;  // N column within strip / A row (M) within tile
    const int half = lane >> 4;  // K-half for fragments; M-half for C/D rows

    float rmin[MTILES][8], lacc[MTILES][8];
#pragma unroll
    for (int t = 0; t < MTILES; ++t)
#pragma unroll
        for (int r = 0; r < 8; ++r) { rmin[t][r] = __builtin_inff(); lacc[t][r] = 0.f; }

    for (int cc = 0; cc < N_COLS / 128; ++cc) {
        const int n = cc * 128 + wid * 16 + c16;   // global class column for this lane
        const __bf16* bhp = Bh + (size_t)n * DIM;
        const __bf16* blp = Bl + (size_t)n * DIM;
        v8f acc[MTILES];
#pragma unroll
        for (int t = 0; t < MTILES; ++t) acc[t] = (v8f)0.0f;

#pragma unroll 2
        for (int ks = 0; ks < DIM / 32; ++ks) {
            // B 32x16 bf16: lane holds col N=c16; elems j -> K=half*16+j (contiguous 32B)
            v16bf bh, bl;
            const u32x4* p = (const u32x4*)(bhp + ks * 32 + half * 16);
            ((u32x4*)&bh)[0] = p[0];
            ((u32x4*)&bh)[1] = p[1];
            const u32x4* q = (const u32x4*)(blp + ks * 32 + half * 16);
            ((u32x4*)&bl)[0] = q[0];
            ((u32x4*)&bl)[1] = q[1];
#pragma unroll
            for (int t = 0; t < MTILES; ++t) {
                const int ar = t * 16 + c16;
                // A 16x32 bf16: lane row M=c16; elems j -> K=half*16+j (contiguous)
                v16bf ah, al;
                ((u32x4*)&ah)[0] = *(const u32x4*)&sAh[ar][ks * 32 + half * 16];
                ((u32x4*)&ah)[1] = *(const u32x4*)&sAh[ar][ks * 32 + half * 16 + 8];
                ((u32x4*)&al)[0] = *(const u32x4*)&sAl[ar][ks * 32 + half * 16];
                ((u32x4*)&al)[1] = *(const u32x4*)&sAl[ar][ks * 32 + half * 16 + 8];
                // split-precision fp32-accurate product: hh + hl + lh
                acc[t] = __builtin_amdgcn_wmma_f32_16x16x32_bf16(false, ah, false, bh, (short)0, acc[t], false, false);
                acc[t] = __builtin_amdgcn_wmma_f32_16x16x32_bf16(false, ah, false, bl, (short)0, acc[t], false, false);
                acc[t] = __builtin_amdgcn_wmma_f32_16x16x32_bf16(false, al, false, bh, (short)0, acc[t], false, false);
            }
        }
        // fused epilogue: d2 -> dist -> label mask -> running min
        const float e2v = e2[n];
#pragma unroll
        for (int t = 0; t < MTILES; ++t) {
#pragma unroll
            for (int r = 0; r < 8; ++r) {
                const int m = t * 16 + r + 8 * half;   // D-matrix: VGPR r holds M = r + 8*(lane>=16)
                float d2   = sX2[m] + e2v - 2.0f * acc[t][r];
                float dist = sqrtf(fmaxf(d2, 0.0f));
                if (n == sLb[m]) { lacc[t][r] += dist; dist = __builtin_inff(); }
                rmin[t][r] = fminf(rmin[t][r], dist);
            }
        }
    }

    // reduce across the 16 lanes that share each row (stay within 16-lane half)
#pragma unroll
    for (int t = 0; t < MTILES; ++t) {
#pragma unroll
        for (int r = 0; r < 8; ++r) {
#pragma unroll
            for (int m = 8; m >= 1; m >>= 1) {
                rmin[t][r] = fminf(rmin[t][r], __shfl_xor(rmin[t][r], m, 32));
                lacc[t][r] += __shfl_xor(lacc[t][r], m, 32);
            }
        }
    }
    if (c16 == 0) {
#pragma unroll
        for (int t = 0; t < MTILES; ++t)
#pragma unroll
            for (int r = 0; r < 8; ++r) {
                const int m = t * 16 + r + 8 * half;
                sMin[wid][m] = rmin[t][r];
                sLab[wid][m] = lacc[t][r];
            }
    }
    __syncthreads();

    if (tid < MROWS) {
        float mn = __builtin_inff(), lb = 0.f;
#pragma unroll
        for (int w = 0; w < 8; ++w) {
            mn = fminf(mn, sMin[w][tid]);
            lb += sLab[w][tid];
        }
        float contrib = (1.0f + lb - mn) * (1.0f / (float)N_ROWS);
        atomicAdd(out, contrib);
    }
}

extern "C" void kernel_launch(void* const* d_in, const int* in_sizes, int n_in,
                              void* d_out, int out_size, void* d_ws, size_t ws_size,
                              hipStream_t stream) {
    const float* WO    = (const float*)d_in[0];
    const float* emb   = (const float*)d_in[1];
    const int*   label = (const int*)d_in[2];
    float*       out   = (float*)d_out;

    char* ws = (char*)d_ws;
    __bf16* Bh = (__bf16*)ws;                                     // C*D bf16
    __bf16* Bl = (__bf16*)(ws + (size_t)N_COLS * DIM * 2);        // C*D bf16
    float*  e2 = (float*)(ws + (size_t)N_COLS * DIM * 4);         // C fp32

    hipMemsetAsync(d_out, 0, sizeof(float), stream);
    prep_emb<<<N_COLS, 256, 0, stream>>>(emb, Bh, Bl, e2);
    dist_main<<<N_ROWS / MROWS, 256, 0, stream>>>(WO, label, Bh, Bl, e2, out);
}
